// Sheaf_Conv_fixed_11089605558415
// MI455X (gfx1250) — matrix operand order repaired and checked by reading, not compiled, not run on hardware
//
#include <hip/hip_runtime.h>

// Sheaf convolution for MI455X (gfx1250, wave32, WMMA).
// All heavy math runs through v_wmma_f32_16x16x32_f16 (f32 accumulate).
// d_ws needs >= 196,608 bytes (f16 weight tiles).

typedef _Float16 h16;
typedef __attribute__((ext_vector_type(8)))  _Float16 v8h;
typedef __attribute__((ext_vector_type(16))) _Float16 v16h;
typedef __attribute__((ext_vector_type(8)))  float    v8f;

#define DIMX 64
#define DIMY 16
#define WAVES_PER_BLOCK 4

struct SheafParams {
  const float* adj;
  const float* emb;
  const int*   ei;        // [2*E] int32: u = ei[e], v = ei[E+e]
  const float* user_op;   // [1024]
  const float* item_op;   // [1024]
  const float* bias[8];
  const h16*   wt;        // f16 weight tiles in d_ws
  float*       out;       // [N, 64]
  int N;
  int E;
  int tiles;
  int woff[8];            // per-layer tile base offsets (in halfs)
};

// ---------------- WMMA helpers ----------------

__device__ __forceinline__ v8f wmma_f16(v16h a, v16h b, v8f c) {
  // D = A(16x32 f16) * B(32x16 f16) + C(16x16 f32)
  return __builtin_amdgcn_wmma_f32_16x16x32_f16(
      /*neg_a=*/false, a, /*neg_b=*/false, b,
      /*c_mod=*/(short)0, c, /*reuse_a=*/false, /*reuse_b=*/false);
}

// A fragment (16x32, M x K) from row-major LDS activations [16][C] (halfs).
// ISA layout: lane<16: row=lane, K {0..7,16..23}; lane>=16: row=lane-16, K {8..15,24..31}.
__device__ __forceinline__ v16h load_afrag(const h16* buf, int C, int kt, int lane) {
  int m = lane & 15, g = lane >> 4;
  const h16* p = buf + m * C + kt * 32 + g * 8;
  v8h lo = *(const v8h*)p;          // k = 32kt + 8g .. +7
  v8h hi = *(const v8h*)(p + 16);   // k = 32kt + 16 + 8g .. +7
  return __builtin_shufflevector(lo, hi, 0,1,2,3,4,5,6,7,8,9,10,11,12,13,14,15);
}

// B fragment: pre-tiled lane-major in d_ws; one contiguous 32B load per lane.
__device__ __forceinline__ v16h load_bfrag(const h16* base, int NT, int kt, int nt, int lane) {
  return *(const v16h*)(base + ((kt * NT + nt) * 32 + lane) * 16);
}

// D fragment -> LDS row-major f16 activations [16][64] with bias + ReLU.
// ISA C/D layout: lane holds column n = 16nt + (lane&15); VGPR r holds row m = r + 8*(lane>>4).
__device__ __forceinline__ void store_act(h16* out, v8f acc, const float* bias, int nt, int lane) {
  int ln = lane & 15, g = lane >> 4;
  int n = nt * 16 + ln;
  float b = bias[n];
  #pragma unroll
  for (int r = 0; r < 8; ++r) {
    float x = acc[r] + b;
    x = x > 0.f ? x : 0.f;
    out[(r + 8 * g) * 64 + n] = (h16)x;
  }
}

__device__ __forceinline__ void gemm_hidden(const h16* in, h16* out, const h16* wt,
                                            const float* bias, int lane) {
  #pragma unroll
  for (int nt = 0; nt < 4; ++nt) {
    v8f acc = {};
    #pragma unroll
    for (int kt = 0; kt < 2; ++kt)
      acc = wmma_f16(load_afrag(in, 64, kt, lane), load_bfrag(wt, 4, kt, nt, lane), acc);
    store_act(out, acc, bias, nt, lane);
  }
}

// ---------------- kernels ----------------

__global__ void zero_out_kernel(float* __restrict__ out, int n) {
  int i = blockIdx.x * blockDim.x + threadIdx.x;
  if (i < n) out[i] = 0.f;
}

// f32 row-major W[K][Nout] -> f16 B-tiles (lane-major 32x16 tiles).
__global__ void prep_weights(const float* __restrict__ W, h16* __restrict__ dst,
                             int K, int Nout) {
  int idx = blockIdx.x * blockDim.x + threadIdx.x;
  if (idx >= K * Nout) return;
  int n = idx % Nout, k = idx / Nout;
  int kt = k >> 5, kl = k & 31, g = (kl >> 3) & 1, j = (kl & 7) | ((kl >> 4) << 3);
  int nt = n >> 4, nl = n & 15;
  int lane = g * 16 + nl;
  int NT = Nout >> 4;
  dst[(size_t)((kt * NT + nt) * 32 + lane) * 16 + j] = (h16)W[(size_t)k * Nout + n];
}

__global__ void __launch_bounds__(WAVES_PER_BLOCK * 32)
sheaf_conv_kernel(SheafParams P) {
  __shared__ h16   sX[WAVES_PER_BLOCK][16 * 128];  // concat(e_u, e_v) f16, persists
  __shared__ h16   sY[WAVES_PER_BLOCK][16 * 64];   // activation ping
  __shared__ h16   sZ[WAVES_PER_BLOCK][16 * 64];   // activation pong
  __shared__ float sHv[WAVES_PER_BLOCK][16 * 16];  // h_v_ [edge][y]
  __shared__ int   sU[WAVES_PER_BLOCK][16];
  __shared__ int   sV[WAVES_PER_BLOCK][16];
  __shared__ float sAW[WAVES_PER_BLOCK][16];       // adj[u,v] weights

  const int wv   = threadIdx.x >> 5;
  const int lane = threadIdx.x & 31;
  const int tile = blockIdx.x * WAVES_PER_BLOCK + wv;
  if (tile >= P.tiles) return;   // wave-uniform exit; WMMA below runs with full EXEC

  const int ln = lane & 15, g = lane >> 4;

  // --- load edge tuples + adjacency weights (lanes 0..15) ---
  if (lane < 16) {
    int ge = tile * 16 + lane;
    int u = 0, v = 0; float w = 0.f;
    if (ge < P.E) {
      u = P.ei[ge];
      v = P.ei[P.E + ge];
      w = P.adj[(size_t)u * P.N + v];
    }
    sU[wv][lane] = u; sV[wv][lane] = v; sAW[wv][lane] = w;
  }
  __builtin_amdgcn_wave_barrier();

  // --- gather concat(e_u, e_v) -> f16 LDS [16 rows][128 cols] ---
  for (int idx = lane; idx < 16 * 128; idx += 32) {
    int row = idx >> 7, c = idx & 127;
    int node = (c < 64) ? sU[wv][row] : sV[wv][row];
    sX[wv][idx] = (h16)P.emb[(size_t)node * DIMX + (c & 63)];
  }
  __builtin_amdgcn_wave_barrier();

  const h16*   w8 = P.wt + P.woff[7];
  const float* b8 = P.bias[7];
  float hacc[32];                 // h_v accumulators [j][r]
  #pragma unroll
  for (int i = 0; i < 32; ++i) hacc[i] = 0.f;

  // pass 0: MLP(concat(e_v,e_u)) -> smat_vu -> h_v_ ; pass 1: smat_uv -> h_v
  for (int pass = 0; pass < 2; ++pass) {
    const bool vu = (pass == 0);

    // ---- layer 1: 128 -> 64 (vu reuses sX via K-tile rotation by 2 == column swap by 64) ----
    {
      const h16* wt0 = P.wt + P.woff[0];
      #pragma unroll
      for (int nt = 0; nt < 4; ++nt) {
        v8f acc = {};
        #pragma unroll
        for (int kt = 0; kt < 4; ++kt) {
          int akt = vu ? ((kt + 2) & 3) : kt;
          acc = wmma_f16(load_afrag(sX[wv], 128, akt, lane),
                         load_bfrag(wt0, 4, kt, nt, lane), acc);
        }
        store_act(sY[wv], acc, P.bias[0], nt, lane);
      }
    }

    // ---- layers 2..7: 64 -> 64, ping-pong Y/Z (odd count -> final result in Y) ----
    gemm_hidden(sY[wv], sZ[wv], P.wt + P.woff[1], P.bias[1], lane);
    gemm_hidden(sZ[wv], sY[wv], P.wt + P.woff[2], P.bias[2], lane);
    gemm_hidden(sY[wv], sZ[wv], P.wt + P.woff[3], P.bias[3], lane);
    gemm_hidden(sZ[wv], sY[wv], P.wt + P.woff[4], P.bias[4], lane);
    gemm_hidden(sY[wv], sZ[wv], P.wt + P.woff[5], P.bias[5], lane);
    gemm_hidden(sZ[wv], sY[wv], P.wt + P.woff[6], P.bias[6], lane);

    // ---- layer 8 (64 -> 1024) fused with the sheaf contractions ----
    if (vu) {
      // h_v_[e,y] = sum_x (mlp8 + bias + op_vu)[e, y*64+x] * e_v[e,x]
      #pragma unroll 1
      for (int y = 0; y < 16; ++y) {
        float s[8];
        #pragma unroll
        for (int r = 0; r < 8; ++r) s[r] = 0.f;
        #pragma unroll
        for (int j = 0; j < 4; ++j) {
          int nt = y * 4 + j;
          v8f acc = {};
          #pragma unroll
          for (int kt = 0; kt < 2; ++kt)
            acc = wmma_f16(load_afrag(sY[wv], 64, kt, lane),
                           load_bfrag(w8, 64, kt, nt, lane), acc);
          int f = nt * 16 + ln;            // flat index y*64+x
          int x = f & 63;
          float bb = b8[f];
          float uo = P.user_op[f], io = P.item_op[f];
          #pragma unroll
          for (int r = 0; r < 8; ++r) {
            int m  = r + 8 * g;
            int ge = tile * 16 + m;
            float opv = (ge == 0) ? io : uo;   // smat_vu: edge 0 += item, rest += user (sep==1)
            float sm  = acc[r] + bb + opv;
            float ev  = (float)sX[wv][m * 128 + 64 + x];
            s[r] += sm * ev;
          }
        }
        #pragma unroll
        for (int off = 1; off < 16; off <<= 1) {
          #pragma unroll
          for (int r = 0; r < 8; ++r) s[r] += __shfl_xor(s[r], off, 32);
        }
        if (ln == 0) {
          #pragma unroll
          for (int r = 0; r < 8; ++r) sHv[wv][(r + 8 * g) * 16 + y] = s[r];
        }
      }
    } else {
      // h_v[e,x] = sum_y (mlp8 + bias + op_uv)[e, y*64+x] * h_v_[e,y]
      #pragma unroll 1
      for (int y = 0; y < 16; ++y) {
        float hb[8];
        #pragma unroll
        for (int r = 0; r < 8; ++r) hb[r] = sHv[wv][(r + 8 * g) * 16 + y];
        #pragma unroll
        for (int j = 0; j < 4; ++j) {
          int nt = y * 4 + j;
          v8f acc = {};
          #pragma unroll
          for (int kt = 0; kt < 2; ++kt)
            acc = wmma_f16(load_afrag(sY[wv], 64, kt, lane),
                           load_bfrag(w8, 64, kt, nt, lane), acc);
          int f = nt * 16 + ln;
          float bb = b8[f];
          float uo = P.user_op[f], io = P.item_op[f];
          #pragma unroll
          for (int r = 0; r < 8; ++r) {
            int m  = r + 8 * g;
            int ge = tile * 16 + m;
            float opv = (ge == 0) ? uo : io;   // smat_uv: edge 0 += user, rest += item
            hacc[j * 8 + r] += (acc[r] + bb + opv) * hb[r];
          }
        }
      }
    }
  }

  // --- c_v = w * h_v, segment-sum over u via f32 atomics ---
  #pragma unroll
  for (int j = 0; j < 4; ++j) {
    int x = j * 16 + ln;
    #pragma unroll
    for (int r = 0; r < 8; ++r) {
      int m  = r + 8 * g;
      int ge = tile * 16 + m;
      if (ge < P.E) {
        float val = sAW[wv][m] * hacc[j * 8 + r];
        __hip_atomic_fetch_add(&P.out[(size_t)sU[wv][m] * DIMX + x], val,
                               __ATOMIC_RELAXED, __HIP_MEMORY_SCOPE_AGENT);
      }
    }
  }
}

// ---------------- host launch ----------------

extern "C" void kernel_launch(void* const* d_in, const int* in_sizes, int n_in,
                              void* d_out, int out_size, void* d_ws, size_t ws_size,
                              hipStream_t stream) {
  (void)n_in; (void)ws_size;
  SheafParams P;
  P.adj     = (const float*)d_in[0];
  P.emb     = (const float*)d_in[1];
  P.ei      = (const int*)  d_in[2];
  P.user_op = (const float*)d_in[3];
  P.item_op = (const float*)d_in[4];
  P.wt      = (const h16*)  d_ws;
  P.out     = (float*)d_out;
  P.E       = in_sizes[2] / 2;
  P.N       = in_sizes[1] / DIMX;
  P.tiles   = (P.E + 15) / 16;

  static const int Ks[8] = {2 * DIMX, 64, 64, 64, 64, 64, 64, 64};
  static const int Ns[8] = {64, 64, 64, 64, 64, 64, 64, DIMY * DIMX};
  int off = 0;
  for (int l = 0; l < 8; ++l) { P.woff[l] = off; off += Ks[l] * Ns[l]; }

  // 1) convert weights f32 -> f16 B-tiles in d_ws (L2-resident thereafter)
  for (int l = 0; l < 8; ++l) {
    const float* W = (const float*)d_in[5 + 2 * l];
    P.bias[l]      = (const float*)d_in[6 + 2 * l];
    int tot = Ks[l] * Ns[l];
    prep_weights<<<(tot + 255) / 256, 256, 0, stream>>>(W, (h16*)d_ws + P.woff[l],
                                                        Ks[l], Ns[l]);
  }

  // 2) zero the output accumulator
  zero_out_kernel<<<(out_size + 255) / 256, 256, 0, stream>>>((float*)d_out, out_size);

  // 3) fused sheaf-conv: 1 wave per 16-edge tile, 4 waves per block
  int blocks = (P.tiles + WAVES_PER_BLOCK - 1) / WAVES_PER_BLOCK;
  sheaf_conv_kernel<<<blocks, WAVES_PER_BLOCK * 32, 0, stream>>>(P);
}
